// neuralGNN_38740605010497
// MI455X (gfx1250) — compile-verified, matrix operand order checked by reading, not run on hardware
//
#include <hip/hip_runtime.h>
#include <math.h>

#define NN      50000
#define NE      800000
#define DF      64      // feature dim D
#define KC      128     // concat dim 2D (GEMM1 K, GEMM2 K via H)
#define HH      128     // hidden
#define NL      3
#define SS      1024
#define SH1     512
#define SH2     256

typedef __attribute__((ext_vector_type(16))) _Float16 v16h;
typedef __attribute__((ext_vector_type(8)))  _Float16 v8h;
typedef __attribute__((ext_vector_type(8)))  float    v8f;
typedef __attribute__((ext_vector_type(4)))  int      v4i;

// ---- CDNA5 async global->LDS copy (ASYNCcnt), with portable fallback ----
#if defined(__has_builtin)
#  if __has_builtin(__builtin_amdgcn_global_load_async_to_lds_b128)
#    define HAVE_ASYNC_B128 1
#  endif
#  if __has_builtin(__builtin_amdgcn_s_wait_asynccnt)
#    define HAVE_WAIT_ASYNC 1
#  endif
#endif

__device__ __forceinline__ void copy_b128_to_lds(const void* g, void* l) {
#if HAVE_ASYNC_B128
  __builtin_amdgcn_global_load_async_to_lds_b128((v4i*)g, (v4i*)l,
                                                 /*offset=*/0, /*cpol=*/0);
#else
  *(v8h*)l = *(const v8h*)g;
#endif
}

__device__ __forceinline__ void async_wait_all() {
#if HAVE_ASYNC_B128
#  if HAVE_WAIT_ASYNC
  __builtin_amdgcn_s_wait_asynccnt(0);
#  else
  asm volatile("s_wait_asynccnt 0x0" ::: "memory");
#  endif
#endif
}

// ---- fragment loaders (CDNA5 ISA 7.12.2 layouts, wave32) ----
// A-matrix 16x32 f16: lane holds row M=lane%16; kbase=(lane/16)*8;
// frag elems [0..7] = K kbase+0..7 ; [8..15] = K kbase+16..23.
// B-matrix 32x16 mirrors this with lane%16 = output column N.
__device__ __forceinline__ v16h load_frag(const _Float16* __restrict__ base,
                                          int stride, int mn0, int k0, int lane) {
  const int mn = mn0 + (lane & 15);
  const int kb = k0 + ((lane >> 4) << 3);
  const _Float16* p = base + mn * stride + kb;
  union { v16h v; v8h h[2]; } u;
  u.h[0] = *(const v8h*)(p);
  u.h[1] = *(const v8h*)(p + 16);
  return u.v;
}

__device__ __forceinline__ v8f wmma_f16(v16h a, v16h b, v8f c) {
  return __builtin_amdgcn_wmma_f32_16x16x32_f16(
      /*neg_a=*/false, a, /*neg_b=*/false, b,
      /*c_mod=*/(short)0, c, /*reuse_a=*/false, /*reuse_b=*/false);
}

// ---------------- phase 0: A = scatter_mean(edge_attr, src) ----------------
__global__ void zero_kernel(float* __restrict__ p, int n) {
  int i = blockIdx.x * blockDim.x + threadIdx.x;
  if (i < n) p[i] = 0.0f;
}

__global__ void edge_count_kernel(const int* __restrict__ src, float* __restrict__ cnt) {
  int e = blockIdx.x * blockDim.x + threadIdx.x;
  if (e < NE) atomicAdd(&cnt[src[e]], 1.0f);
}

__global__ void edge_scatter_kernel(const int* __restrict__ src,
                                    const float* __restrict__ ea,
                                    float* __restrict__ Asum) {
  int tid = blockIdx.x * blockDim.x + threadIdx.x;
  if (tid >= NE * 16) return;
  int e = tid >> 4, q = tid & 15;
  int s = src[e];
  float4 v = *(const float4*)(ea + (size_t)e * DF + q * 4);
  float* p = Asum + (size_t)s * DF + q * 4;
  atomicAdd(p + 0, v.x); atomicAdd(p + 1, v.y);
  atomicAdd(p + 2, v.z); atomicAdd(p + 3, v.w);
}

__global__ void finalize_A_kernel(const float* __restrict__ cnt, float* __restrict__ A) {
  int i = blockIdx.x * blockDim.x + threadIdx.x;
  if (i < NN * DF) A[i] /= fmaxf(cnt[i >> 6], 1.0f);
}

// ---------------- weight prep: f32 -> f16, transposed to [out][k] ----------------
__global__ void prep_weights_kernel(const float* __restrict__ pw1,
                                    const float* __restrict__ pw2,
                                    _Float16* __restrict__ W1t,
                                    _Float16* __restrict__ W2t) {
  int tid = blockIdx.x * blockDim.x + threadIdx.x;
  if (tid < NL * HH * KC) {            // W1t[l][h][k] = pw1[l][k][h]
    int l = tid / (HH * KC), rem = tid % (HH * KC);
    int h = rem / KC, k = rem % KC;
    W1t[tid] = (_Float16)pw1[((size_t)l * KC + k) * HH + h];
  }
  if (tid < NL * DF * KC) {            // W2t[l][d][k] = pw2[l][k][d]
    int l = tid / (DF * KC), rem = tid % (DF * KC);
    int d = rem / KC, k = rem % KC;
    W2t[tid] = (_Float16)pw2[((size_t)l * HH + k) * DF + d];
  }
}

// ---------------- fused processor layer: concat -> GEMM1 -> relu -> GEMM2 -> LN ----------------
__global__ __launch_bounds__(256) void gnn_layer_kernel(
    const float* __restrict__ xin, const float* __restrict__ Aagg,
    const _Float16* __restrict__ W1t, const _Float16* __restrict__ W2t,
    const float* __restrict__ b1, const float* __restrict__ b2,
    const float* __restrict__ g, const float* __restrict__ beta,
    float* __restrict__ xout) {
  __shared__ __align__(16) _Float16 sT[128 * KC];   // 32 KB: concat input, then Y1
  __shared__ __align__(16) _Float16 sW[HH * KC];    // 32 KB: W1, then W2 (first 16 KB)

  const int tid  = threadIdx.x;
  const int lane = tid & 31;
  const int wave = tid >> 5;
  const int R0   = blockIdx.x * 128;
  const int m0   = wave * 16;
  const int half = lane >> 4;
  const int ln   = lane & 15;

  // kick off W1 staging into LDS (async path uses ASYNCcnt)
  for (int i = tid; i < (HH * KC) / 8; i += 256)
    copy_b128_to_lds((const v8h*)W1t + i, (v8h*)sW + i);

  // stage concat([x, x*A]) as f16 into LDS: 128 rows x 128 cols
  for (int i = tid; i < 128 * 16; i += 256) {
    int r = i >> 4, k4 = (i & 15) << 2;
    int row = R0 + r;
    float4 xv = make_float4(0.f, 0.f, 0.f, 0.f);
    float4 av = make_float4(0.f, 0.f, 0.f, 0.f);
    if (row < NN) {
      xv = *(const float4*)(xin + (size_t)row * DF + k4);
      av = *(const float4*)(Aagg + (size_t)row * DF + k4);
    }
    _Float16* p = sT + r * KC + k4;
    p[0] = (_Float16)xv.x;            p[1] = (_Float16)xv.y;
    p[2] = (_Float16)xv.z;            p[3] = (_Float16)xv.w;
    p[DF + 0] = (_Float16)(xv.x * av.x); p[DF + 1] = (_Float16)(xv.y * av.y);
    p[DF + 2] = (_Float16)(xv.z * av.z); p[DF + 3] = (_Float16)(xv.w * av.w);
  }
  async_wait_all();
  __syncthreads();

  // ---- GEMM1: [16 x 128] x [128 x 128] per wave; A and B both from LDS ----
  v8f acc[8];
#pragma unroll
  for (int t = 0; t < 8; ++t) acc[t] = (v8f){0.f,0.f,0.f,0.f,0.f,0.f,0.f,0.f};
#pragma unroll
  for (int kk = 0; kk < 4; ++kk) {
    const int k0 = kk * 32;
    v16h a = load_frag(sT, KC, m0, k0, lane);
#pragma unroll
    for (int t = 0; t < 8; ++t) {
      v16h b = load_frag(sW, KC, t * 16, k0, lane);
      acc[t] = wmma_f16(a, b, acc[t]);
    }
  }
  __syncthreads();   // all waves done reading sT / sW

  // overlap: start async W2 staging while writing Y1
  for (int i = tid; i < (DF * KC) / 8; i += 256)
    copy_b128_to_lds((const v8h*)W2t + i, (v8h*)sW + i);

  // bias + relu -> Y1 (f16) into reused LDS
#pragma unroll
  for (int t = 0; t < 8; ++t) {
    float bv = b1[t * 16 + ln];
#pragma unroll
    for (int r = 0; r < 8; ++r) {
      float v = acc[t][r] + bv;
      v = v > 0.f ? v : 0.f;
      sT[(m0 + r + half * 8) * KC + t * 16 + ln] = (_Float16)v;
    }
  }
  async_wait_all();
  __syncthreads();

  // ---- GEMM2: [16 x 128] x [128 x 64] per wave ----
  v8f acc2[4];
#pragma unroll
  for (int t = 0; t < 4; ++t) acc2[t] = (v8f){0.f,0.f,0.f,0.f,0.f,0.f,0.f,0.f};
#pragma unroll
  for (int kk = 0; kk < 4; ++kk) {
    const int k0 = kk * 32;
    v16h a = load_frag(sT, KC, m0, k0, lane);
#pragma unroll
    for (int t = 0; t < 4; ++t) {
      v16h b = load_frag(sW, KC, t * 16, k0, lane);
      acc2[t] = wmma_f16(a, b, acc2[t]);
    }
  }

  float gv[4], bev[4];
#pragma unroll
  for (int t = 0; t < 4; ++t) {
    float bv = b2[t * 16 + ln];
#pragma unroll
    for (int r = 0; r < 8; ++r) acc2[t][r] += bv;
    gv[t]  = g[t * 16 + ln];
    bev[t] = beta[t * 16 + ln];
  }

  // rowwise LayerNorm over 64 features; row lives in one 16-lane half
#pragma unroll
  for (int r = 0; r < 8; ++r) {
    float s = 0.f, q = 0.f;
#pragma unroll
    for (int t = 0; t < 4; ++t) { float v = acc2[t][r]; s += v; q += v * v; }
#pragma unroll
    for (int off = 8; off >= 1; off >>= 1) {
      s += __shfl_xor(s, off, 32);
      q += __shfl_xor(q, off, 32);
    }
    float mean = s * (1.0f / DF);
    float var  = q * (1.0f / DF) - mean * mean;
    float rs   = rsqrtf(var + 1e-5f);
    int row = R0 + m0 + r + half * 8;
    if (row < NN) {
#pragma unroll
      for (int t = 0; t < 4; ++t) {
        int col = t * 16 + ln;
        xout[(size_t)row * DF + col] = (acc2[t][r] - mean) * rs * gv[t] + bev[t];
      }
    }
  }
}

// ---------------- final: LN(1) collapses t to tbeta; supernode MLP; softmax([1,1])=1 ----------------
__global__ __launch_bounds__(512) void supernode_kernel(
    const float* __restrict__ sw1, const float* __restrict__ sb1,
    const float* __restrict__ sw2, const float* __restrict__ sb2,
    const float* __restrict__ sw3, const float* __restrict__ sb3,
    const float* __restrict__ tbeta, float* __restrict__ out) {
  __shared__ float z1[SH1];
  __shared__ float z2[SH2];
  __shared__ float red[SH1];
  int j = threadIdx.x;
  float t = tbeta[0];           // LayerNorm over a size-1 dim => beta exactly
  float s = 0.f;
  for (int i = 0; i < SS; ++i) s += sw1[(size_t)i * SH1 + j];   // z row is constant t
  z1[j] = fmaxf(t * s + sb1[j], 0.f);
  __syncthreads();
  if (j < SH2) {
    float s2 = 0.f;
    for (int i = 0; i < SH1; ++i) s2 += z1[i] * sw2[(size_t)i * SH2 + j];
    z2[j] = fmaxf(s2 + sb2[j], 0.f);
  }
  __syncthreads();
  red[j] = (j < SH2) ? z2[j] * sw3[j] : 0.f;
  __syncthreads();
  for (int st = 256; st >= 1; st >>= 1) {
    if (j < st) red[j] += red[j + st];
    __syncthreads();
  }
  if (j == 0) {
    float z3 = red[0] + sb3[0];
    out[0] = expf(z3) / expf(z3);   // softmax over a [1,1] tensor
  }
}

extern "C" void kernel_launch(void* const* d_in, const int* in_sizes, int n_in,
                              void* d_out, int out_size, void* d_ws, size_t ws_size,
                              hipStream_t stream) {
  const float* x     = (const float*)d_in[0];
  const int*   ei    = (const int*)d_in[1];     // src = ei[0..NE)
  const float* ea    = (const float*)d_in[2];
  const float* pw1   = (const float*)d_in[4];
  const float* pb1   = (const float*)d_in[5];
  const float* pw2   = (const float*)d_in[6];
  const float* pb2   = (const float*)d_in[7];
  const float* pg    = (const float*)d_in[8];
  const float* pbeta = (const float*)d_in[9];
  const float* tbeta = (const float*)d_in[15];
  const float* sw1   = (const float*)d_in[16];
  const float* sb1   = (const float*)d_in[17];
  const float* sw2   = (const float*)d_in[18];
  const float* sb2   = (const float*)d_in[19];
  const float* sw3   = (const float*)d_in[20];
  const float* sb3   = (const float*)d_in[21];
  float* out = (float*)d_out;

  // workspace carve-up (all 16B aligned)
  float* Af   = (float*)d_ws;                       // NN*DF
  float* cnt  = Af + (size_t)NN * DF;               // NN
  float* xbuf = cnt + NN;                           // NN*DF   (NN*4 bytes is 16B-multiple)
  _Float16* W1t = (_Float16*)(xbuf + (size_t)NN * DF);  // NL*HH*KC
  _Float16* W2t = W1t + NL * HH * KC;                   // NL*DF*KC

  const int T = 256;
  zero_kernel<<<(NN * DF + T - 1) / T, T, 0, stream>>>(Af, NN * DF);
  zero_kernel<<<(NN + T - 1) / T, T, 0, stream>>>(cnt, NN);
  prep_weights_kernel<<<(NL * HH * KC + T - 1) / T, T, 0, stream>>>(pw1, pw2, W1t, W2t);
  edge_count_kernel<<<(NE + T - 1) / T, T, 0, stream>>>(ei, cnt);
  edge_scatter_kernel<<<(NE * 16 + T - 1) / T, T, 0, stream>>>(ei, ea, Af);
  finalize_A_kernel<<<(NN * DF + T - 1) / T, T, 0, stream>>>(cnt, Af);

  const int nblk = (NN + 127) / 128;
  for (int l = 0; l < NL; ++l) {
    gnn_layer_kernel<<<nblk, 256, 0, stream>>>(
        l == 0 ? x : xbuf, Af,
        W1t + (size_t)l * HH * KC, W2t + (size_t)l * DF * KC,
        pb1 + l * HH, pb2 + l * DF, pg + l * DF, pbeta + l * DF, xbuf);
  }

  supernode_kernel<<<1, 512, 0, stream>>>(sw1, sb1, sw2, sb2, sw3, sb3, tbeta, out);
}
